// EnsembleRSSM_40664750358702
// MI455X (gfx1250) — matrix operand order repaired and checked by reading, not compiled, not run on hardware
//
#include <hip/hip_runtime.h>
#include <cstdint>
#include <cstddef>

#define BATCH 256
#define TT    64
#define EMB   1536
#define ACT_S 6
#define NSTOCH 32
#define NDETER 1024
#define NHID  1024
#define OUTC  2240   // 6*32 + 2*1024 per (b,t)

typedef __attribute__((ext_vector_type(8)))  float  v8f;
typedef __attribute__((ext_vector_type(16))) __bf16 v16bf;
typedef __attribute__((ext_vector_type(8)))  __bf16 v8bf;

struct Frag { v8bf lo, hi; };

__device__ __forceinline__ v16bf fcat(Frag f) {
  union { v16bf v; v8bf h[2]; } u;
  u.h[0] = f.lo; u.h[1] = f.hi;
  return u.v;
}

__device__ __forceinline__ float eluf(float x)   { return x > 0.f ? x : expm1f(x); }
__device__ __forceinline__ float sigf(float x)   { return 1.f / (1.f + __expf(-x)); }
__device__ __forceinline__ float splusf(float x) { return x > 20.f ? x : log1pf(__expf(x)); }

// ---------------------------------------------------------------------------
// Pipelined WMMA core: one wave computes a 32(M) x 64(N) strip of A[M,K]*Bt[N,K]^T.
// Register double-buffering: next k-step's 12 fragment loads are issued before
// the current k-step's 8 WMMAs, so global-load latency overlaps matrix issue.
// ---------------------------------------------------------------------------
__device__ __forceinline__ void gemm_core_32x64(
    const __bf16* __restrict__ A, const __bf16* __restrict__ Bt,
    int K, int m0, int n0, int lane, v8f (&acc)[2][4]) {
  const int row = lane & 15, kh = lane >> 4;
  const v8f vz = {0.f, 0.f, 0.f, 0.f, 0.f, 0.f, 0.f, 0.f};
  #pragma unroll
  for (int mi = 0; mi < 2; ++mi)
    #pragma unroll
    for (int j = 0; j < 4; ++j) acc[mi][j] = vz;

  const __bf16* ap[2];
  #pragma unroll
  for (int mi = 0; mi < 2; ++mi)
    ap[mi] = A + (size_t)(m0 + mi * 16 + row) * K;       // A row base
  const __bf16* bp[4];
  #pragma unroll
  for (int j = 0; j < 4; ++j)
    bp[j] = Bt + (size_t)(n0 + j * 16 + row) * K + kh * 16;  // B col base (+lane K half)

  // A frag order used here (contiguous K, same as B): lanes 0-15 hold K {0..15}, lanes 16-31 {16..31}
  auto ldA = [&](int mi, int k0) {
    Frag f;
    f.lo = *reinterpret_cast<const v8bf*>(ap[mi] + k0 + kh * 16);
    f.hi = *reinterpret_cast<const v8bf*>(ap[mi] + k0 + kh * 16 + 8);
    return f;
  };
  // B frag order used here (same K order as A): lane = col, 16 consecutive K starting at kh*16
  auto ldB = [&](int j, int k0) {
    Frag f;
    f.lo = *reinterpret_cast<const v8bf*>(bp[j] + k0);
    f.hi = *reinterpret_cast<const v8bf*>(bp[j] + k0 + 8);
    return f;
  };

  Frag a_c[2], b_c[4];
  #pragma unroll
  for (int mi = 0; mi < 2; ++mi) a_c[mi] = ldA(mi, 0);
  #pragma unroll
  for (int j = 0; j < 4; ++j)    b_c[j]  = ldB(j, 0);

  for (int k0 = 0; k0 < K; k0 += 32) {
    const int k1 = (k0 + 32 < K) ? k0 + 32 : 0;  // wrap: unconditional valid prefetch
    Frag a_n[2], b_n[4];
    #pragma unroll
    for (int mi = 0; mi < 2; ++mi) a_n[mi] = ldA(mi, k1);
    #pragma unroll
    for (int j = 0; j < 4; ++j)    b_n[j]  = ldB(j, k1);
    #pragma unroll
    for (int mi = 0; mi < 2; ++mi)
      #pragma unroll
      for (int j = 0; j < 4; ++j)
        acc[mi][j] = __builtin_amdgcn_wmma_f32_16x16x32_bf16(
            false, fcat(a_c[mi]), false, fcat(b_c[j]), (short)0, acc[mi][j],
            false, false);
    #pragma unroll
    for (int mi = 0; mi < 2; ++mi) a_c[mi] = a_n[mi];
    #pragma unroll
    for (int j = 0; j < 4; ++j)    b_c[j]  = b_n[j];
  }
}

// ---------------------------------------------------------------------------
// Weight convert: fp32 [K][N] -> bf16 [N][K]  (B-operand friendly layout)
// ---------------------------------------------------------------------------
__global__ __launch_bounds__(256) void rssm_wconv(const float* __restrict__ W,
                                                  __bf16* __restrict__ Wt,
                                                  int K, int N) {
  int idx = blockIdx.x * 256 + threadIdx.x;
  if (idx >= K * N) return;
  int k = idx / N, n = idx % N;
  Wt[(size_t)n * K + k] = (__bf16)W[(size_t)k * N + n];
}

// ---------------------------------------------------------------------------
// Step prologue: mask state by is_first, x_img = elu([stoch,a] @ W_img + b),
// build bf16 GRU A-panel [x_img | deter_masked], keep fp32 deter_masked.
// ---------------------------------------------------------------------------
__global__ __launch_bounds__(256) void rssm_prep(
    const float* __restrict__ stoch, const float* __restrict__ deter,
    const float* __restrict__ action, const int* __restrict__ is_first, int t,
    const float* __restrict__ W_img, const float* __restrict__ b_img,
    __bf16* __restrict__ A_gru, float* __restrict__ dmask) {
  int idx = blockIdx.x * 256 + threadIdx.x;
  int r = idx >> 10, c = idx & 1023;
  float m = 1.f - (float)is_first[r * TT + t];
  float acc = b_img[c];
  #pragma unroll 8
  for (int k = 0; k < NSTOCH; ++k)
    acc += (stoch[r * NSTOCH + k] * m) * W_img[k * NHID + c];
  const float* ar = action + ((size_t)r * TT + t) * ACT_S;
  #pragma unroll
  for (int k = 0; k < ACT_S; ++k)
    acc += (ar[k] * m) * W_img[(NSTOCH + k) * NHID + c];
  A_gru[(size_t)r * 2048 + c] = (__bf16)eluf(acc);
  float d = deter[(size_t)r * NDETER + c] * m;
  A_gru[(size_t)r * 2048 + 1024 + c] = (__bf16)d;
  dmask[(size_t)r * NDETER + c] = d;
}

// embed[:,t,:] fp32 -> bf16 into obs A-panel cols [0,1536)
__global__ __launch_bounds__(256) void rssm_embed(const float* __restrict__ embed,
                                                  __bf16* __restrict__ A_obs, int t) {
  int idx = blockIdx.x * 256 + threadIdx.x;  // 256*1536
  int r = idx / EMB, c = idx % EMB;
  A_obs[(size_t)r * 2560 + c] = (__bf16)embed[((size_t)r * TT + t) * EMB + c];
}

// ---------------------------------------------------------------------------
// GEMM kernel: 2 waves/block stacked in M (block tile 64x64, B shared in WGP$).
// EPI 0: write fp32 C.  EPI 1: C = elu(acc + bias[col]) -> bf16.
// ---------------------------------------------------------------------------
template <int EPI>
__global__ __launch_bounds__(64) void rssm_gemm(
    const __bf16* __restrict__ A, const __bf16* __restrict__ Bt,
    int K, int N, float* __restrict__ Cf, __bf16* __restrict__ Cb,
    const float* __restrict__ bias) {
  const int lane = threadIdx.x & 31;
  const int wv   = threadIdx.x >> 5;
  const int m0 = blockIdx.x * 64 + wv * 32, n0 = blockIdx.y * 64;
  const int row = lane & 15, kh = lane >> 4;

  v8f acc[2][4];
  gemm_core_32x64(A, Bt, K, m0, n0, lane, acc);

  #pragma unroll
  for (int mi = 0; mi < 2; ++mi) {
    #pragma unroll
    for (int j = 0; j < 4; ++j) {
      const int col = n0 + j * 16 + row;
      #pragma unroll
      for (int e = 0; e < 8; ++e) {
        const int rr = m0 + mi * 16 + kh * 8 + e;  // f32 C/D layout: VGPR e -> M
        float val = acc[mi][j][e];
        if (EPI == 0) {
          Cf[(size_t)rr * N + col] = val;
        } else {
          val += bias[col];
          Cb[(size_t)rr * N + col] = (__bf16)eluf(val);
        }
      }
    }
  }
}

// ---------------------------------------------------------------------------
// Stats GEMM: [256,K] x [64,K]^T, mean/softplus-std epilogue straight to d_out.
// POST=true: posterior (omean/ostd/stoch @ 0/32/64, updates stoch state)
// POST=false: prior (pmean/pstd/pmean @ 1120/1152/1184)
// ---------------------------------------------------------------------------
template <bool POST>
__global__ __launch_bounds__(64) void rssm_gemm_stats(
    const __bf16* __restrict__ A, const __bf16* __restrict__ Bt, int K,
    const float* __restrict__ bias, float* __restrict__ out, int t,
    float* __restrict__ stoch) {
  const int lane = threadIdx.x & 31;
  const int wv   = threadIdx.x >> 5;
  const int m0 = blockIdx.x * 64 + wv * 32;
  const int row = lane & 15, kh = lane >> 4;

  v8f acc[2][4];
  gemm_core_32x64(A, Bt, K, m0, /*n0=*/0, lane, acc);

  #pragma unroll
  for (int mi = 0; mi < 2; ++mi) {
    #pragma unroll
    for (int j = 0; j < 4; ++j) {
      const int col = j * 16 + row;           // 0..63: [0,32)=mean, [32,64)=std_raw
      #pragma unroll
      for (int e = 0; e < 8; ++e) {
        const int rr = m0 + mi * 16 + kh * 8 + e;
        float val = acc[mi][j][e] + bias[col];
        float* ob = out + ((size_t)rr * TT + t) * OUTC;
        if (POST) {
          if (col < 32) { ob[col] = val; ob[64 + col] = val; stoch[rr * NSTOCH + col] = val; }
          else          { ob[col] = splusf(val) + 0.1f; }          // 32+(col-32)==col
        } else {
          if (col < 32) { ob[1120 + col] = val; ob[1184 + col] = val; }
          else          { ob[1120 + col] = splusf(val) + 0.1f; }   // 1152+(col-32)
        }
      }
    }
  }
}

// ---------------------------------------------------------------------------
// GRU update: LN over 3072 (bias added), gates, deter_new fan-out.
// one block per batch row.
// ---------------------------------------------------------------------------
__global__ __launch_bounds__(256) void rssm_gru(
    const float* __restrict__ parts, const float* __restrict__ bg,
    const float* __restrict__ lns, const float* __restrict__ lnb,
    const float* __restrict__ dmask, float* __restrict__ deter,
    __bf16* __restrict__ A_ens, __bf16* __restrict__ A_obs,
    float* __restrict__ out, int t) {
  const int r = blockIdx.x, tid = threadIdx.x;
  const float* __restrict__ p = parts + (size_t)r * 3072;
  float s = 0.f, ss = 0.f;
  for (int i = tid; i < 3072; i += 256) {
    float v = p[i] + bg[i];
    s += v; ss += v * v;
  }
  __shared__ float rs[256], rss[256];
  rs[tid] = s; rss[tid] = ss;
  __syncthreads();
  for (int o = 128; o > 0; o >>= 1) {
    if (tid < o) { rs[tid] += rs[tid + o]; rss[tid] += rss[tid + o]; }
    __syncthreads();
  }
  __shared__ float mean_s, rstd_s;
  if (tid == 0) {
    float m = rs[0] * (1.f / 3072.f);
    float var = rss[0] * (1.f / 3072.f) - m * m;
    mean_s = m; rstd_s = rsqrtf(var + 1e-5f);
  }
  __syncthreads();
  const float mean = mean_s, rstd = rstd_s;
  for (int j = tid; j < NDETER; j += 256) {
    float rv = ((p[j]        + bg[j]        - mean) * rstd) * lns[j]        + lnb[j];
    float cv = ((p[1024 + j] + bg[1024 + j] - mean) * rstd) * lns[1024 + j] + lnb[1024 + j];
    float uv = ((p[2048 + j] + bg[2048 + j] - mean) * rstd) * lns[2048 + j] + lnb[2048 + j];
    float reset = sigf(rv);
    float cand  = tanhf(reset * cv);
    float up    = sigf(uv - 1.f);                       // update_bias = -1
    float dn    = up * cand + (1.f - up) * dmask[(size_t)r * NDETER + j];
    deter[(size_t)r * NDETER + j] = dn;
    A_ens[(size_t)r * NDETER + j] = (__bf16)dn;
    A_obs[(size_t)r * 2560 + 1536 + j] = (__bf16)dn;
    float* ob = out + ((size_t)r * TT + t) * OUTC;
    ob[96 + j] = dn;
    ob[1216 + j] = dn;
  }
}

// ---------------------------------------------------------------------------
extern "C" void kernel_launch(void* const* d_in, const int* in_sizes, int n_in,
                              void* d_out, int out_size, void* d_ws, size_t ws_size,
                              hipStream_t stream) {
  const float* embed       = (const float*)d_in[0];
  const float* action      = (const float*)d_in[1];
  const int*   is_first    = (const int*)  d_in[2];
  const float* W_obs       = (const float*)d_in[3];
  const float* b_obs       = (const float*)d_in[4];
  const float* W_obs_stats = (const float*)d_in[5];
  const float* b_obs_stats = (const float*)d_in[6];
  const float* W_img       = (const float*)d_in[7];
  const float* b_img       = (const float*)d_in[8];
  const float* W_gru       = (const float*)d_in[9];
  const float* b_gru       = (const float*)d_in[10];
  const float* lns         = (const float*)d_in[11];
  const float* lnb         = (const float*)d_in[12];
  const float* W_ens1      = (const float*)d_in[13];  // member 0 at offset 0
  const float* b_ens1      = (const float*)d_in[14];
  const float* W_ens2      = (const float*)d_in[15];
  const float* b_ens2      = (const float*)d_in[16];
  float* out = (float*)d_out;

  char* wp = (char*)d_ws;
  auto alloc = [&](size_t bytes) -> void* {
    void* p = (void*)wp;
    wp += (bytes + 255) & ~(size_t)255;
    return p;
  };
  __bf16* Wt_gru = (__bf16*)alloc(2048ull * 3072 * 2);
  __bf16* Wt_obs = (__bf16*)alloc(2560ull * 1024 * 2);
  __bf16* Wt_e1  = (__bf16*)alloc(1024ull * 1024 * 2);
  __bf16* Wt_e2  = (__bf16*)alloc(1024ull * 64 * 2);
  __bf16* Wt_os  = (__bf16*)alloc(1024ull * 64 * 2);
  float*  stochS = (float*) alloc(256ull * 32 * 4);      // contiguous with deterS
  float*  deterS = (float*) alloc(256ull * 1024 * 4);
  float*  dmask  = (float*) alloc(256ull * 1024 * 4);
  __bf16* A_gru  = (__bf16*)alloc(256ull * 2048 * 2);
  float*  parts  = (float*) alloc(256ull * 3072 * 4);
  __bf16* A_ens  = (__bf16*)alloc(256ull * 1024 * 2);
  __bf16* hid    = (__bf16*)alloc(256ull * 1024 * 2);
  __bf16* A_obs  = (__bf16*)alloc(256ull * 2560 * 2);
  __bf16* x_obs  = (__bf16*)alloc(256ull * 1024 * 2);

  auto cdiv = [](int a, int b) { return (a + b - 1) / b; };

  // One-time (per call) weight convert/transpose into bf16 [N][K]
  rssm_wconv<<<cdiv(2048 * 3072, 256), 256, 0, stream>>>(W_gru, Wt_gru, 2048, 3072);
  rssm_wconv<<<cdiv(2560 * 1024, 256), 256, 0, stream>>>(W_obs, Wt_obs, 2560, 1024);
  rssm_wconv<<<cdiv(1024 * 1024, 256), 256, 0, stream>>>(W_ens1, Wt_e1, 1024, 1024);
  rssm_wconv<<<cdiv(1024 * 64, 256), 256, 0, stream>>>(W_ens2, Wt_e2, 1024, 64);
  rssm_wconv<<<cdiv(1024 * 64, 256), 256, 0, stream>>>(W_obs_stats, Wt_os, 1024, 64);
  // zero initial (stoch, deter) state — contiguous in ws
  hipMemsetAsync(stochS, 0, (256ull * 32 + 256ull * 1024) * sizeof(float), stream);

  for (int t = 0; t < TT; ++t) {
    rssm_prep<<<1024, 256, 0, stream>>>(stochS, deterS, action, is_first, t,
                                        W_img, b_img, A_gru, dmask);
    rssm_embed<<<1536, 256, 0, stream>>>(embed, A_obs, t);
    // GRU: [256,2048] x [2048,3072]
    rssm_gemm<0><<<dim3(4, 48), 64, 0, stream>>>(A_gru, Wt_gru, 2048, 3072,
                                                 parts, nullptr, nullptr);
    rssm_gru<<<256, 256, 0, stream>>>(parts, b_gru, lns, lnb, dmask, deterS,
                                      A_ens, A_obs, out, t);
    // ens1: [256,1024] x [1024,1024], elu epilogue
    rssm_gemm<1><<<dim3(4, 16), 64, 0, stream>>>(A_ens, Wt_e1, 1024, 1024,
                                                 nullptr, hid, b_ens1);
    // prior stats: [256,1024] x [1024,64]
    rssm_gemm_stats<false><<<4, 64, 0, stream>>>(hid, Wt_e2, 1024, b_ens2,
                                                 out, t, nullptr);
    // obs: [256,2560] x [2560,1024], elu epilogue
    rssm_gemm<1><<<dim3(4, 16), 64, 0, stream>>>(A_obs, Wt_obs, 2560, 1024,
                                                 nullptr, x_obs, b_obs);
    // posterior stats: [256,1024] x [1024,64] (+ stoch state update)
    rssm_gemm_stats<true><<<4, 64, 0, stream>>>(x_obs, Wt_os, 1024, b_obs_stats,
                                                out, t, stochS);
  }
}
